// Backbone_44994077393320
// MI455X (gfx1250) — matrix operand order repaired and checked by reading, not compiled, not run on hardware
//
#include <hip/hip_runtime.h>
#include <hip/hip_bf16.h>

typedef __bf16 bf16_t;
typedef __attribute__((ext_vector_type(16))) __bf16 v16bf;
typedef __attribute__((ext_vector_type(4)))  __bf16 v4bf;
typedef __attribute__((ext_vector_type(8)))  float  v8f;

#define S_LEN 4096
#define H_DIM 1024
#define FF_DIM 3072
#define G_NUM 100
#define MAX_SPAN 30
#define MAXK 819
#define NPAD 131072   /* 4096 * 32 == 2^17 */
#define MASK_VAL 10000.0f
#define LN_EPS 1e-5f

// ---------------------------------------------------------------------------
// Block reduction helper (deterministic tree reduce in shared memory)
// ---------------------------------------------------------------------------
__device__ __forceinline__ float blockReduceSum(float v, float* sm, int nthreads) {
    int t = threadIdx.x;
    sm[t] = v;
    __syncthreads();
    for (int s = nthreads >> 1; s > 0; s >>= 1) {
        if (t < s) sm[t] += sm[t + s];
        __syncthreads();
    }
    float r = sm[0];
    __syncthreads();
    return r;
}

// ---------------------------------------------------------------------------
// Split-f32 WMMA GEMM: C[M,N] = act(A[M,K] @ B[K,N] + bias[N])
//
// f32 operands are decomposed as x = hi + lo (hi = bf16(x), lo = bf16(x-hi)).
// Accumulate hiA*hiB + loA*hiB + hiA*loB in f32 on the bf16 WMMA pipe:
// error ~2^-16 relative (f32-class) at 3x bf16-WMMA cost, which is free
// against the MI455X matrix ceiling (total ~390 GFLOP-equivalent here).
//
// Block tile 128x128, 8 wave32s, each wave 64x32 via 4x2 WMMA tiles.
// ---------------------------------------------------------------------------
__global__ __launch_bounds__(256)
void gemm_bias_act(const float* __restrict__ A, const float* __restrict__ B,
                   const float* __restrict__ bias, float* __restrict__ C,
                   int M, int N, int K, int applyGelu)
{
    __shared__ __align__(64) bf16_t AsHi[128][32];  // [m][k]  8 KB each
    __shared__ __align__(64) bf16_t AsLo[128][32];
    __shared__ __align__(64) bf16_t BsHi[128][32];  // [n][k] (B transposed)
    __shared__ __align__(64) bf16_t BsLo[128][32];

    const int tid  = threadIdx.x;
    const int lane = tid & 31;
    const int wave = tid >> 5;       // 0..7
    const int wm   = wave >> 2;      // 0..1 -> 64-row group
    const int wn   = wave & 3;       // 0..3 -> 32-col group
    const int laneHalf = lane >> 4;  // 0/1
    const int laneIdx  = lane & 15;

    const int blockM = blockIdx.y * 128;
    const int blockN = blockIdx.x * 128;

    v8f acc[4][2];
#pragma unroll
    for (int i = 0; i < 4; i++)
#pragma unroll
        for (int j = 0; j < 2; j++)
#pragma unroll
            for (int r = 0; r < 8; r++) acc[i][j][r] = 0.0f;

    for (int k0 = 0; k0 < K; k0 += 32) {
        // ---- stage A tile 128x32: vectorized float4 loads, hi/lo split ----
#pragma unroll
        for (int it = 0; it < 4; it++) {
            int idx = it * 256 + tid;          // 0..1023 float4 slots
            int m   = idx >> 3;                // 8 float4 per 32-wide row
            int k4  = (idx & 7) << 2;
            float4 f = *(const float4*)&A[(size_t)(blockM + m) * K + (k0 + k4)];
            float xs[4] = {f.x, f.y, f.z, f.w};
            v4bf hi, lo;
#pragma unroll
            for (int e = 0; e < 4; e++) {
                bf16_t h = (bf16_t)xs[e];
                hi[e] = h;
                lo[e] = (bf16_t)(xs[e] - (float)h);
            }
            *(v4bf*)&AsHi[m][k4] = hi;
            *(v4bf*)&AsLo[m][k4] = lo;
        }
        // ---- stage B tile 32x128 (transposed to [n][k]), float4 loads ----
#pragma unroll
        for (int it = 0; it < 4; it++) {
            int idx = it * 256 + tid;          // 0..1023 float4 slots
            int kk  = idx >> 5;                // 0..31
            int n4  = (idx & 31) << 2;         // 0..124
            float4 f = *(const float4*)&B[(size_t)(k0 + kk) * N + (blockN + n4)];
            float xs[4] = {f.x, f.y, f.z, f.w};
#pragma unroll
            for (int e = 0; e < 4; e++) {
                bf16_t h = (bf16_t)xs[e];
                BsHi[n4 + e][kk] = h;
                BsLo[n4 + e][kk] = (bf16_t)(xs[e] - (float)h);
            }
        }
        __syncthreads();

        // ---- fragments + WMMA ----
        // B layout: elem e -> K = 16*laneHalf + e (contiguous 16)
        v16bf bHi[2], bLo[2];
#pragma unroll
        for (int j = 0; j < 2; j++) {
            const bf16_t* cH = &BsHi[wn * 32 + j * 16 + laneIdx][laneHalf * 16];
            const bf16_t* cL = &BsLo[wn * 32 + j * 16 + laneIdx][laneHalf * 16];
            bHi[j] = *(const v16bf*)cH;
            bLo[j] = *(const v16bf*)cL;
        }
#pragma unroll
        for (int i = 0; i < 4; i++) {
            // A layout: elem e -> K = 16*laneHalf + e (contiguous 16)
            const bf16_t* rH = &AsHi[wm * 64 + i * 16 + laneIdx][0];
            const bf16_t* rL = &AsLo[wm * 64 + i * 16 + laneIdx][0];
            v16bf aHi, aLo;
#pragma unroll
            for (int e = 0; e < 8; e++) {
                aHi[e]     = rH[laneHalf * 16 + e];
                aHi[8 + e] = rH[laneHalf * 16 + 8 + e];
                aLo[e]     = rL[laneHalf * 16 + e];
                aLo[8 + e] = rL[laneHalf * 16 + 8 + e];
            }
#pragma unroll
            for (int j = 0; j < 2; j++) {
                acc[i][j] = __builtin_amdgcn_wmma_f32_16x16x32_bf16(
                    false, aHi, false, bHi[j], (short)0, acc[i][j], false, false);
                acc[i][j] = __builtin_amdgcn_wmma_f32_16x16x32_bf16(
                    false, aLo, false, bHi[j], (short)0, acc[i][j], false, false);
                acc[i][j] = __builtin_amdgcn_wmma_f32_16x16x32_bf16(
                    false, aHi, false, bLo[j], (short)0, acc[i][j], false, false);
            }
        }
        __syncthreads();
    }

    // epilogue: C/D layout: vgpr r, lane l -> M = r + 8*(l/16), N = l%16
#pragma unroll
    for (int i = 0; i < 4; i++) {
#pragma unroll
        for (int j = 0; j < 2; j++) {
            int nG = blockN + wn * 32 + j * 16 + laneIdx;
            float bv = bias[nG];
            int mBase = blockM + wm * 64 + i * 16 + laneHalf * 8;
#pragma unroll
            for (int r = 0; r < 8; r++) {
                float v = acc[i][j][r] + bv;
                if (applyGelu) v = 0.5f * v * (1.0f + erff(v * 0.70710678118654752f));
                C[(size_t)(mBase + r) * N + nG] = v;
            }
        }
    }
}

// ---------------------------------------------------------------------------
// Per-row LayerNorm (+gamma,+beta) and fused logit dot:  reps, logit = reps.w+b
// ---------------------------------------------------------------------------
__global__ __launch_bounds__(256)
void row_ln_logit(const float* __restrict__ h, const float* __restrict__ g,
                  const float* __restrict__ beta, const float* __restrict__ w,
                  const float* __restrict__ b, float* __restrict__ reps,
                  float* __restrict__ logits)
{
    __shared__ float sm[256];
    const int row = blockIdx.x;
    const float* hr = h + (size_t)row * FF_DIM;
    float s = 0.f;
    for (int i = threadIdx.x; i < FF_DIM; i += 256) s += hr[i];
    float mu = blockReduceSum(s, sm, 256) * (1.0f / FF_DIM);
    float v = 0.f;
    for (int i = threadIdx.x; i < FF_DIM; i += 256) { float d = hr[i] - mu; v += d * d; }
    float var = blockReduceSum(v, sm, 256) * (1.0f / FF_DIM);
    float rstd = rsqrtf(var + LN_EPS);
    float dot = 0.f;
    float* rr = reps + (size_t)row * FF_DIM;
    for (int i = threadIdx.x; i < FF_DIM; i += 256) {
        float rv = (hr[i] - mu) * rstd * g[i] + beta[i];
        rr[i] = rv;
        dot += rv * w[i];
    }
    dot = blockReduceSum(dot, sm, 256);
    if (threadIdx.x == 0) logits[row] = dot + b[0];
}

// ---------------------------------------------------------------------------
// Band joint: for each row i, spans 0..29:  clip(temp[i].end_reps[j] + sl + el)
// band stored [S][32] (slots 30,31 unused)
// ---------------------------------------------------------------------------
__global__ __launch_bounds__(256)
void band_joint(const float* __restrict__ temp, const float* __restrict__ ereps,
                const float* __restrict__ sl, const float* __restrict__ el,
                float* __restrict__ band)
{
    __shared__ float trow[FF_DIM];
    __shared__ float sm[256];
    const int i = blockIdx.x;
    for (int t = threadIdx.x; t < FF_DIM; t += 256) trow[t] = temp[(size_t)i * FF_DIM + t];
    __syncthreads();
    for (int span = 0; span < MAX_SPAN; span++) {
        int j = i + span;
        float dot = 0.f;
        if (j < S_LEN) {
            const float* er = ereps + (size_t)j * FF_DIM;
            for (int t = threadIdx.x; t < FF_DIM; t += 256) dot += trow[t] * er[t];
        }
        dot = blockReduceSum(dot, sm, 256);
        if (threadIdx.x == 0) {
            if (j < S_LEN) {
                float v = dot + sl[i] + el[j];
                v = fminf(fmaxf(v, -MASK_VAL), MASK_VAL);
                band[i * 32 + span] = v;
            } else {
                band[i * 32 + span] = -3.0e38f;
            }
        }
        __syncthreads();
    }
}

// ---------------------------------------------------------------------------
// Gold flags / costs
// ---------------------------------------------------------------------------
__global__ void zero_flags(int* __restrict__ flags)
{
    int t = blockIdx.x * blockDim.x + threadIdx.x;
    if (t < NPAD) flags[t] = 0;
}

__global__ void set_flags(const int* __restrict__ gold, int* __restrict__ flags)
{
    int t = blockIdx.x * blockDim.x + threadIdx.x;
    if (t < G_NUM) {
        int gs = gold[2 * t], ge = gold[2 * t + 1];
        flags[gs * 32 + (ge - gs)] = 1;
    }
}

__global__ __launch_bounds__(128)
void gold_cost(const float* __restrict__ band, const int* __restrict__ gold,
               float* __restrict__ acc)
{
    __shared__ float sm[128];
    int t = threadIdx.x;
    float lg = 0.f;
    if (t < G_NUM) {
        int gs = gold[2 * t], ge = gold[2 * t + 1];
        float l = band[gs * 32 + (ge - gs)];
        // log(sigmoid(l)), numerically stable
        float ls = (l >= 0.f) ? (-log1pf(expf(-l))) : (l - log1pf(expf(l)));
        lg = fmaxf(ls, -100.0f);
    }
    lg = blockReduceSum(lg, sm, 128);
    if (t == 0) acc[0] = lg;
}

__global__ __launch_bounds__(256)
void junk_cost(const float* __restrict__ band, const int* __restrict__ flags,
               float* __restrict__ pj, float* __restrict__ pc)
{
    __shared__ float sm[256];
    int t = blockIdx.x * blockDim.x + threadIdx.x;
    float term = 0.f, cnt = 0.f;
    int i = t >> 5, span = t & 31, j = i + span;
    if (span < MAX_SPAN && j < S_LEN) {
        cnt = 1.f;                       // mask count includes gold positions
        if (!flags[t]) {
            float l = band[t];
            // log(1 - sigmoid(l)) = log(sigmoid(-l)), stable
            float ls = (l > 0.f) ? (-l - log1pf(expf(-l))) : (-log1pf(expf(l)));
            term = fmaxf(ls, -100.0f);
        }
    }
    float ts = blockReduceSum(term, sm, 256);
    float cs = blockReduceSum(cnt, sm, 256);
    if (threadIdx.x == 0) { pj[blockIdx.x] = ts; pc[blockIdx.x] = cs; }
}

__global__ __launch_bounds__(512)
void final_cost(const float* __restrict__ pj, const float* __restrict__ pc,
                const float* __restrict__ acc, float* __restrict__ out)
{
    __shared__ float s1[512], s2[512];
    int t = threadIdx.x;
    s1[t] = pj[t]; s2[t] = pc[t];
    __syncthreads();
    for (int s = 256; s > 0; s >>= 1) {
        if (t < s) { s1[t] += s1[t + s]; s2[t] += s2[t + s]; }
        __syncthreads();
    }
    if (t == 0) out[0] = -(acc[0] / (float)G_NUM) - (s1[0] / s2[0]);
}

// ---------------------------------------------------------------------------
// Top-k: fill key/idx, global bitonic sort (descending), finalize 819 indices
// ---------------------------------------------------------------------------
__global__ void fill_sort(const float* __restrict__ band,
                          float* __restrict__ keys, int* __restrict__ idxs)
{
    int t = blockIdx.x * blockDim.x + threadIdx.x;
    if (t < NPAD) {
        int i = t >> 5, span = t & 31, j = i + span;
        bool valid = (span < MAX_SPAN) && (j < S_LEN);
        keys[t] = valid ? band[t] : -3.0e38f;
        idxs[t] = valid ? (i * S_LEN + j) : 0x7FFFFFFF;
    }
}

__global__ void bitonic_step(float* __restrict__ key, int* __restrict__ idx,
                             int j, int k)
{
    int t = blockIdx.x * blockDim.x + threadIdx.x;
    int p = t ^ j;
    if (p > t && t < NPAD) {
        float a = key[t], b = key[p];
        int ia = idx[t], ib = idx[p];
        bool desc = ((t & k) == 0);
        bool a_first = (a > b) || (a == b && ia < ib);   // value desc, idx asc tiebreak
        if (desc ? !a_first : a_first) {
            key[t] = b; key[p] = a; idx[t] = ib; idx[p] = ia;
        }
    }
}

__global__ __launch_bounds__(1024)
void topk_finalize(const int* __restrict__ idxs, float* __restrict__ out)
{
    __shared__ int s[1024];
    int t = threadIdx.x;
    s[t] = (t < MAXK) ? idxs[t] : 0x7FFFFFFF;
    __syncthreads();
    for (int k = 2; k <= 1024; k <<= 1) {
        for (int j = k >> 1; j > 0; j >>= 1) {
            int p = t ^ j;
            if (p > t) {
                bool up = ((t & k) == 0);
                int a = s[t], b = s[p];
                if (up ? (a > b) : (a < b)) { s[t] = b; s[p] = a; }
            }
            __syncthreads();
        }
    }
    if (t < MAXK) {
        int idx = s[t];
        out[t]            = (float)(idx >> 12);     // idx / 4096
        out[MAXK + t]     = (float)(idx & 4095);    // idx % 4096
        out[2 * MAXK + t] = 1.0f;
    }
}

// ---------------------------------------------------------------------------
// Launcher
// ---------------------------------------------------------------------------
extern "C" void kernel_launch(void* const* d_in, const int* in_sizes, int n_in,
                              void* d_out, int out_size, void* d_ws, size_t ws_size,
                              hipStream_t stream)
{
    const float* seq    = (const float*)d_in[0];
    const int*   gold   = (const int*)d_in[2];
    const float* W_sm   = (const float*)d_in[3];
    const float* b_sm   = (const float*)d_in[4];
    const float* g_sm   = (const float*)d_in[5];
    const float* be_sm  = (const float*)d_in[6];
    const float* W_em   = (const float*)d_in[7];
    const float* b_em   = (const float*)d_in[8];
    const float* g_em   = (const float*)d_in[9];
    const float* be_em  = (const float*)d_in[10];
    const float* w_st   = (const float*)d_in[11];
    const float* b_st   = (const float*)d_in[12];
    const float* w_en   = (const float*)d_in[13];
    const float* b_en   = (const float*)d_in[14];
    const float* W_s2e  = (const float*)d_in[15];
    const float* b_s2e  = (const float*)d_in[16];
    float* out = (float*)d_out;

    const size_t MAT = (size_t)S_LEN * FF_DIM;          // 12,582,912 floats
    const size_t NEED = 3 * MAT + (size_t)S_LEN * 32 + 3 * (size_t)NPAD +
                        2 * (size_t)S_LEN + 1032;
    if (ws_size < NEED * sizeof(float)) return;

    float* ws    = (float*)d_ws;
    size_t o     = 0;
    float* h     = ws + o; o += MAT;       // gelu output, later reused for `temp`
    float* sreps = ws + o; o += MAT;
    float* ereps = ws + o; o += MAT;
    float* band  = ws + o; o += (size_t)S_LEN * 32;
    float* keys  = ws + o; o += NPAD;
    int*   idxs  = (int*)(ws + o); o += NPAD;
    int*   flags = (int*)(ws + o); o += NPAD;
    float* slog  = ws + o; o += S_LEN;
    float* elog  = ws + o; o += S_LEN;
    float* pj    = ws + o; o += 512;
    float* pc    = ws + o; o += 512;
    float* acc   = ws + o; o += 8;

    dim3 gFF(FF_DIM / 128, S_LEN / 128);   // (24, 32)

    // start branch: gelu(X@W_sm + b) -> h ; LN -> sreps, start logits
    gemm_bias_act<<<gFF, 256, 0, stream>>>(seq, W_sm, b_sm, h, S_LEN, FF_DIM, H_DIM, 1);
    row_ln_logit<<<S_LEN, 256, 0, stream>>>(h, g_sm, be_sm, w_st, b_st, sreps, slog);
    // end branch
    gemm_bias_act<<<gFF, 256, 0, stream>>>(seq, W_em, b_em, h, S_LEN, FF_DIM, H_DIM, 1);
    row_ln_logit<<<S_LEN, 256, 0, stream>>>(h, g_em, be_em, w_en, b_en, ereps, elog);
    // temp = sreps @ W_s2e + b_s2e  (reuse h)
    gemm_bias_act<<<gFF, 256, 0, stream>>>(sreps, W_s2e, b_s2e, h, S_LEN, FF_DIM, FF_DIM, 0);
    // band joint logits (clipped)
    band_joint<<<S_LEN, 256, 0, stream>>>(h, ereps, slog, elog, band);

    // costs
    zero_flags<<<NPAD / 256, 256, 0, stream>>>(flags);
    set_flags<<<1, 128, 0, stream>>>(gold, flags);
    gold_cost<<<1, 128, 0, stream>>>(band, gold, acc);
    junk_cost<<<512, 256, 0, stream>>>(band, flags, pj, pc);
    final_cost<<<1, 512, 0, stream>>>(pj, pc, acc, out + 3 * MAXK + (size_t)S_LEN * H_DIM);

    // top-k over the band (all top-819 values live in the band)
    fill_sort<<<NPAD / 256, 256, 0, stream>>>(band, keys, idxs);
    for (int k = 2; k <= NPAD; k <<= 1)
        for (int j = k >> 1; j > 0; j >>= 1)
            bitonic_step<<<NPAD / 256, 256, 0, stream>>>(keys, idxs, j, k);
    topk_finalize<<<1, 1024, 0, stream>>>(idxs, out);

    // sequence_output passthrough
    hipMemcpyAsync(out + 3 * MAXK, seq, (size_t)S_LEN * H_DIM * sizeof(float),
                   hipMemcpyDeviceToDevice, stream);
}
